// MatryoshkaAttention_23184233463861
// MI455X (gfx1250) — matrix operand order repaired and checked by reading, not compiled, not run on hardware
//
#include <hip/hip_runtime.h>
#include <hip/hip_bf16.h>
#include <math.h>

#define USE_ASYNC_LDS 1   // gfx1250 global_load_async_to_lds_b128 path

// ---------------- types ----------------
typedef _Float16 v16h __attribute__((ext_vector_type(16)));
typedef _Float16 v8h  __attribute__((ext_vector_type(8)));
typedef float    v8f  __attribute__((ext_vector_type(8)));

struct __align__(16) H8 { _Float16 v[8]; };
struct __align__(8)  H4 { _Float16 v[4]; };

// problem constants (reference: B=2, T=4096, D=1024, active_dim=512)
#define BB   2
#define TT   4096
#define DD   1024
#define AD   512
#define NH   8
#define HD   64

// Build a 16-half WMMA fragment from two aligned 8-half chunks.
__device__ inline v16h ld_frag(const _Float16* p0, const _Float16* p1) {
    v8h lo = *(const v8h*)p0;
    v8h hi = *(const v8h*)p1;
    v16h r;
#pragma unroll
    for (int i = 0; i < 8; ++i) { r[i] = lo[i]; r[8 + i] = hi[i]; }
    return r;
}

__device__ inline void load4(const float* p, float o[4]) {
    float4 v = *(const float4*)p;
    o[0] = v.x; o[1] = v.y; o[2] = v.z; o[3] = v.w;
}
__device__ inline void load4(const _Float16* p, float o[4]) {
    H4 v = *(const H4*)p;
#pragma unroll
    for (int i = 0; i < 4; ++i) o[i] = (float)v.v[i];
}

#if USE_ASYNC_LDS
// Async global->LDS 16-byte copy (per-lane addresses), tracked by ASYNCcnt.
// Low 32 bits of a generic LDS pointer are the wave's LDS byte offset.
__device__ inline void async_copy_b128(_Float16* lds_dst, const _Float16* gsrc) {
    unsigned loff = (unsigned)(unsigned long long)(uintptr_t)lds_dst;
    unsigned long long gaddr = (unsigned long long)(uintptr_t)gsrc;
    asm volatile("global_load_async_to_lds_b128 %0, %1, off"
                 :: "v"(loff), "v"(gaddr) : "memory");
}
__device__ inline void async_wait0() {
    asm volatile("s_wait_asynccnt 0x0" ::: "memory");
}
#endif

// ---------------------------------------------------------------------------
// GEMM: C[M,N] = A[M,K] * W[N,K]^T     (W always fp32; A/C templated)
// block tile 128(M) x 64(N), BK=32, 256 threads = 8 waves, wave tile 32x32.
// ---------------------------------------------------------------------------
template <typename TIN, typename TOUT>
__global__ __launch_bounds__(256)
void matry_gemm_atb(const TIN* __restrict__ A, int lda,
                    const float* __restrict__ W, int ldw,
                    TOUT* __restrict__ C, int ldc, int Ksize) {
    __shared__ __align__(16) _Float16 sA[128 * 40];  // 32 K + 8 pad per row
    __shared__ __align__(16) _Float16 sW[64 * 40];

    const int tid   = threadIdx.x;
    const int wave  = tid >> 5;
    const int lane  = tid & 31;
    const int n_l   = lane & 15;
    const int s_l   = lane >> 4;
    const int mw    = wave >> 1;     // 0..3
    const int nw    = wave & 1;      // 0..1
    const int rowB  = blockIdx.y * 128;
    const int colB  = blockIdx.x * 64;

    v8f acc[2][2];
#pragma unroll
    for (int a = 0; a < 2; ++a)
#pragma unroll
        for (int b = 0; b < 2; ++b)
#pragma unroll
            for (int i = 0; i < 8; ++i) acc[a][b][i] = 0.0f;

    for (int k0 = 0; k0 < Ksize; k0 += 32) {
        __syncthreads();
        // stage A tile: 128x32 -> 1024 chunks of 4
#pragma unroll
        for (int t = 0; t < 4; ++t) {
            int c   = tid + t * 256;
            int row = c >> 3, col = (c & 7) * 4;
            float tmp[4];
            load4(A + (size_t)(rowB + row) * lda + k0 + col, tmp);
#pragma unroll
            for (int i = 0; i < 4; ++i) sA[row * 40 + col + i] = (_Float16)tmp[i];
            if (k0 + 32 < Ksize)
                __builtin_prefetch(A + (size_t)(rowB + row) * lda + k0 + 32 + col, 0, 1);
        }
        // stage W tile: 64x32 -> 512 chunks of 4
#pragma unroll
        for (int t = 0; t < 2; ++t) {
            int c   = tid + t * 256;
            int row = c >> 3, col = (c & 7) * 4;
            float tmp[4];
            load4(W + (size_t)(colB + row) * ldw + k0 + col, tmp);
#pragma unroll
            for (int i = 0; i < 4; ++i) sW[row * 40 + col + i] = (_Float16)tmp[i];
            if (k0 + 32 < Ksize)
                __builtin_prefetch(W + (size_t)(colB + row) * ldw + k0 + 32 + col, 0, 1);
        }
        __syncthreads();

        v16h aF[2], bF[2];
#pragma unroll
        for (int ms = 0; ms < 2; ++ms) {
            const _Float16* rp = &sA[(mw * 32 + ms * 16 + n_l) * 40];
            aF[ms] = ld_frag(rp + 8 * s_l, rp + 16 + 8 * s_l);
        }
#pragma unroll
        for (int ns = 0; ns < 2; ++ns) {
            const _Float16* rp = &sW[(nw * 32 + ns * 16 + n_l) * 40];
            bF[ns] = ld_frag(rp + 8 * s_l, rp + 16 + 8 * s_l);
        }
#pragma unroll
        for (int ms = 0; ms < 2; ++ms)
#pragma unroll
            for (int ns = 0; ns < 2; ++ns)
                acc[ms][ns] = __builtin_amdgcn_wmma_f32_16x16x32_f16(
                    false, aF[ms], false, bF[ns], (short)0, acc[ms][ns], false, false);
    }

#pragma unroll
    for (int ms = 0; ms < 2; ++ms)
#pragma unroll
        for (int ns = 0; ns < 2; ++ns)
#pragma unroll
            for (int r = 0; r < 8; ++r) {
                int row = rowB + mw * 32 + ms * 16 + 8 * s_l + r;
                int col = colB + nw * 32 + ns * 16 + n_l;
                C[(size_t)row * ldc + col] = (TOUT)acc[ms][ns][r];
            }
}

// ---------------------------------------------------------------------------
// Flash attention (causal), head_dim=64, f16 in / f16 out, fp32 accum.
// block = (b, h, 64-query tile), 128 threads = 4 waves x 16 query rows.
// K tile staged via gfx1250 async global->LDS DMA; V staged transposed.
// ---------------------------------------------------------------------------
__global__ __launch_bounds__(128)
void matry_flash_attn(const _Float16* __restrict__ Qh,
                      const _Float16* __restrict__ Kh,
                      const _Float16* __restrict__ Vh,
                      _Float16* __restrict__ Oh) {
    __shared__ __align__(16) _Float16 sK[32 * 72];   // [key][d], 64+8 pad
    __shared__ __align__(16) _Float16 sVt[64 * 40];  // [d][key], 32+8 pad
    __shared__ __align__(16) _Float16 sP[4 * 16 * 40];

    const int tid  = threadIdx.x;
    const int wave = tid >> 5;
    const int lane = tid & 31;
    const int n_l  = lane & 15;
    const int s_l  = lane >> 4;
    const int q0   = blockIdx.x * 64;
    const int h    = blockIdx.y;
    const int b    = blockIdx.z;
    const int wqb  = q0 + wave * 16;                 // wave's query base

    // Q fragments (loop-invariant): A-layout straight from global
    v16h aq[2];
    {
        const _Float16* base =
            Qh + (size_t)(b * TT + wqb + n_l) * AD + h * HD;
#pragma unroll
        for (int kc = 0; kc < 2; ++kc)
            aq[kc] = ld_frag(base + kc * 32 + 8 * s_l,
                             base + kc * 32 + 16 + 8 * s_l);
    }

    v8f o[4];
    float m_i[8], l_i[8];
#pragma unroll
    for (int t = 0; t < 4; ++t)
#pragma unroll
        for (int i = 0; i < 8; ++i) o[t][i] = 0.0f;
#pragma unroll
    for (int r = 0; r < 8; ++r) { m_i[r] = -1e30f; l_i[r] = 0.0f; }

    const float scale = 0.125f;                      // 1/sqrt(64)
    _Float16* sPw = &sP[wave * 16 * 40];
    const int kend = q0 + 64;

    for (int k0 = 0; k0 < kend; k0 += 32) {
        __syncthreads();
        // stage K tile [32][64] (async DMA) and V tile transposed [64][32]
#pragma unroll
        for (int t = 0; t < 2; ++t) {
            int c = tid + t * 128;
            int kr = c >> 3, col = (c & 7) * 8;
            const size_t src = (size_t)(b * TT + k0 + kr) * AD + h * HD + col;
#if USE_ASYNC_LDS
            async_copy_b128(&sK[kr * 72 + col], Kh + src);
#else
            H8 kv = *(const H8*)(Kh + src);
            *(H8*)(&sK[kr * 72 + col]) = kv;
#endif
            H8 vv = *(const H8*)(Vh + src);
#pragma unroll
            for (int i = 0; i < 8; ++i) sVt[(col + i) * 40 + kr] = vv.v[i];
            if (k0 + 32 < kend) {
                __builtin_prefetch(Kh + src + 32 * AD, 0, 1);
                __builtin_prefetch(Vh + src + 32 * AD, 0, 1);
            }
        }
#if USE_ASYNC_LDS
        async_wait0();   // ASYNCcnt==0 before signaling the barrier
#endif
        __syncthreads();

        // S = Q * K^T  (two 16-key column tiles)
        v8f s0, s1;
#pragma unroll
        for (int i = 0; i < 8; ++i) { s0[i] = 0.0f; s1[i] = 0.0f; }
#pragma unroll
        for (int kc = 0; kc < 2; ++kc) {
            const _Float16* r0 = &sK[(n_l) * 72 + kc * 32 + 8 * s_l];
            const _Float16* r1 = &sK[(16 + n_l) * 72 + kc * 32 + 8 * s_l];
            v16h b0 = ld_frag(r0, r0 + 16);
            v16h b1 = ld_frag(r1, r1 + 16);
            s0 = __builtin_amdgcn_wmma_f32_16x16x32_f16(false, aq[kc], false, b0,
                                                        (short)0, s0, false, false);
            s1 = __builtin_amdgcn_wmma_f32_16x16x32_f16(false, aq[kc], false, b1,
                                                        (short)0, s1, false, false);
        }
#pragma unroll
        for (int r = 0; r < 8; ++r) { s0[r] *= scale; s1[r] *= scale; }

        // causal mask (only tiles that can touch the diagonal)
        if (k0 + 31 > wqb) {
            const int key0 = k0 + n_l, key1 = k0 + 16 + n_l;
#pragma unroll
            for (int r = 0; r < 8; ++r) {
                int qrow = wqb + 8 * s_l + r;
                if (key0 > qrow) s0[r] = -1e30f;
                if (key1 > qrow) s1[r] = -1e30f;
            }
        }

        // online softmax per row (row r+8s lives in one 16-lane half)
        v8f p0, p1;
#pragma unroll
        for (int r = 0; r < 8; ++r) {
            float mx = fmaxf(s0[r], s1[r]);
            mx = fmaxf(mx, __shfl_xor(mx, 1));
            mx = fmaxf(mx, __shfl_xor(mx, 2));
            mx = fmaxf(mx, __shfl_xor(mx, 4));
            mx = fmaxf(mx, __shfl_xor(mx, 8));
            float mnew  = fmaxf(m_i[r], mx);
            float alpha = __expf(m_i[r] - mnew);
            m_i[r] = mnew;
            p0[r] = __expf(s0[r] - mnew);
            p1[r] = __expf(s1[r] - mnew);
            float rs = p0[r] + p1[r];
            rs += __shfl_xor(rs, 1);
            rs += __shfl_xor(rs, 2);
            rs += __shfl_xor(rs, 4);
            rs += __shfl_xor(rs, 8);
            l_i[r] = l_i[r] * alpha + rs;
#pragma unroll
            for (int t = 0; t < 4; ++t) o[t][r] *= alpha;
        }

        // C-layout P -> A-layout via per-wave LDS patch
#pragma unroll
        for (int r = 0; r < 8; ++r) {
            sPw[(8 * s_l + r) * 40 + n_l]      = (_Float16)p0[r];
            sPw[(8 * s_l + r) * 40 + 16 + n_l] = (_Float16)p1[r];
        }
        const _Float16* pr = &sPw[n_l * 40];
        v16h ap = ld_frag(pr + 8 * s_l, pr + 16 + 8 * s_l);

        // O += P * V   (4 head-dim column tiles)
#pragma unroll
        for (int t = 0; t < 4; ++t) {
            const _Float16* vr = &sVt[(t * 16 + n_l) * 40 + 8 * s_l];
            v16h bv = ld_frag(vr, vr + 16);
            o[t] = __builtin_amdgcn_wmma_f32_16x16x32_f16(false, ap, false, bv,
                                                          (short)0, o[t], false, false);
        }
    }

    // normalize and store O (f16)
#pragma unroll
    for (int r = 0; r < 8; ++r) {
        float inv = 1.0f / l_i[r];
        size_t row = (size_t)(b * TT + wqb + 8 * s_l + r) * AD + h * HD;
#pragma unroll
        for (int t = 0; t < 4; ++t)
            Oh[row + t * 16 + n_l] = (_Float16)(o[t][r] * inv);
    }
}

__global__ void matry_zero_out(float4* __restrict__ p) {
    p[(size_t)blockIdx.x * blockDim.x + threadIdx.x] = make_float4(0.f, 0.f, 0.f, 0.f);
}

// ---------------------------------------------------------------------------
extern "C" void kernel_launch(void* const* d_in, const int* in_sizes, int n_in,
                              void* d_out, int out_size, void* d_ws, size_t ws_size,
                              hipStream_t stream) {
    (void)in_sizes; (void)n_in; (void)out_size; (void)ws_size;
    const float* x  = (const float*)d_in[0];
    const float* wq = (const float*)d_in[1];
    const float* wk = (const float*)d_in[2];
    const float* wv = (const float*)d_in[3];
    const float* wo = (const float*)d_in[4];
    float* out = (float*)d_out;

    const size_t NQ = (size_t)BB * TT * AD;          // 4,194,304 halves
    _Float16* Qh = (_Float16*)d_ws;
    _Float16* Kh = Qh + NQ;
    _Float16* Vh = Kh + NQ;
    _Float16* Oh = Vh + NQ;

    // 1) zero full padded output (B*T*D floats)
    matry_zero_out<<<(BB * TT * DD / 4) / 256, 256, 0, stream>>>((float4*)out);

    // 2-4) Q/K/V projections: [8192,512] = x[:, :512] @ W[:512,:512]^T (f16 out)
    dim3 gP(AD / 64, (BB * TT) / 128);
    matry_gemm_atb<float, _Float16><<<gP, 256, 0, stream>>>(x, DD, wq, DD, Qh, AD, AD);
    matry_gemm_atb<float, _Float16><<<gP, 256, 0, stream>>>(x, DD, wk, DD, Kh, AD, AD);
    matry_gemm_atb<float, _Float16><<<gP, 256, 0, stream>>>(x, DD, wv, DD, Vh, AD, AD);

    // 5) causal flash attention, 8 heads of dim 64
    dim3 gA(TT / 64, NH, BB);
    matry_flash_attn<<<gA, 128, 0, stream>>>(Qh, Kh, Vh, Oh);

    // 6) output projection into d_out cols 0..511 (f32, row stride D=1024)
    matry_gemm_atb<_Float16, float><<<gP, 256, 0, stream>>>(Oh, AD, wo, DD, out, DD, AD);
}
